// GRUAutoencoder_53541062312021
// MI455X (gfx1250) — matrix operand order repaired and checked by reading, not compiled, not run on hardware
//
#include <hip/hip_runtime.h>
#include <hip/hip_bf16.h>

// GRU autoencoder for MI455X (gfx1250), wave32 + v_wmma_f32_16x16x32_f16.
// B=8192 T=100 F=H=64, 3H=192 gate rows (r,z,n).
#define B_ 8192
#define T_ 100
#define F_ 64
#define H_ 64
#define G_ 192
#define GRU_WAVES 2   // waves per block; each wave owns 16 batch rows for all T

typedef _Float16 v8h  __attribute__((ext_vector_type(8)));
typedef _Float16 v16h __attribute__((ext_vector_type(16)));
typedef float    v8f  __attribute__((ext_vector_type(8)));

union FragU { v16h v; v8h h[2]; };

// A-matrix fragment (16x32 f16). ISA layout: lane holds row M=lane&15;
// lanes 0-15: K = {0..7, 16..23}; lanes 16-31: K = {8..15, 24..31} (per k-step of 32).
__device__ __forceinline__ v16h loadA(const _Float16* row, int kstep, int khalf) {
  FragU u;
  const int o = kstep * 32 + khalf * 8;
  u.h[0] = *(const v8h*)(row + o);
  u.h[1] = *(const v8h*)(row + o + 16);
  return u.v;
}

// B-matrix fragment (32x16 f16) from row-major W[192][64] in LDS.
// Lane holds column N=lane&15 of the B tile == row (nt*16 + lane&15) of W;
// lanes 0-15 hold K={0..7,16..23}, lanes 16-31 hold K={8..15,24..31} (matches A).
__device__ __forceinline__ v16h loadB(const _Float16* wrow, int kstep, int khalf) {
  FragU u;
  const int o = kstep * 32 + khalf * 8;
  u.h[0] = *(const v8h*)(wrow + o);
  u.h[1] = *(const v8h*)(wrow + o + 16);
  return u.v;
}

__device__ __forceinline__ v8f wmma(v16h a, v16h b, v8f c) {
  return __builtin_amdgcn_wmma_f32_16x16x32_f16(false, a, false, b, (short)0, c, false, false);
}

__device__ __forceinline__ float sigm(float x) { return 1.0f / (1.0f + __expf(-x)); }
__device__ __forceinline__ float tanh_fast(float x) {
  float e = __expf(-2.0f * x);
  return (1.0f - e) / (1.0f + e);
}

// One GRU layer. Input seq is f16 with arbitrary (bstride, tstride) so the
// decoder's broadcast context (tstride=0) reuses the same kernel.
// Each wave: 16 batch rows, full T loop. Gates = X*Wih^T + h*Whh^T done as
// 12 N-tiles of 16 gates, K=64 in two WMMA k-steps. The compiler hoists all
// weight B-fragments into registers (loop-invariant), so the steady-state
// loop touches LDS only for the recurrent h tile.
__global__ __launch_bounds__(GRU_WAVES * 32)
void gru_layer_kernel(const _Float16* __restrict__ in, long long in_bstride, long long in_tstride,
                      const float* __restrict__ Wih, const float* __restrict__ Whh,
                      const float* __restrict__ bih, const float* __restrict__ bhh,
                      const _Float16* __restrict__ h0,       // [B,H] f16 or nullptr (zeros)
                      _Float16* __restrict__ yseq,           // [B,T,H] f16
                      _Float16* __restrict__ hT)             // [B,H]  f16
{
  __shared__ _Float16 sWih[G_ * H_];
  __shared__ _Float16 sWhh[G_ * H_];
  __shared__ float    sbih[G_];
  __shared__ float    sbhh[G_];
  __shared__ _Float16 sh[GRU_WAVES][16 * H_];   // per-wave h tile, f16

  const int tid = threadIdx.x;
  for (int i = tid; i < G_ * H_; i += GRU_WAVES * 32) {
    sWih[i] = (_Float16)Wih[i];
    sWhh[i] = (_Float16)Whh[i];
  }
  for (int i = tid; i < G_; i += GRU_WAVES * 32) { sbih[i] = bih[i]; sbhh[i] = bhh[i]; }
  __syncthreads();

  const int lane  = tid & 31;
  const int wave  = tid >> 5;
  const int c     = lane & 15;   // N / column index this lane owns in C/D
  const int khalf = lane >> 4;   // lane half selects K sub-chunks / M offset
  const int b0    = blockIdx.x * (GRU_WAVES * 16) + wave * 16;

  // Per-lane gate biases (loop invariant). r/z use bih+bhh fused; n keeps them split.
  float brz_r[4], brz_z[4], b_in[4], b_hn[4];
#pragma unroll
  for (int q = 0; q < 4; ++q) {
    int j = q * 16 + c;
    brz_r[q] = sbih[j] + sbhh[j];
    brz_z[q] = sbih[64 + j] + sbhh[64 + j];
    b_in[q]  = sbih[128 + j];
    b_hn[q]  = sbhh[128 + j];
  }

  // h kept in registers in C-layout (lane: M = v+8*khalf, N = q*16+c) and
  // mirrored in wave-private LDS (A-layout source for the recurrent matmul).
  float hreg[4][8];
  _Float16* shw = sh[wave];
#pragma unroll
  for (int q = 0; q < 4; ++q)
#pragma unroll
    for (int v = 0; v < 8; ++v) {
      int m = v + 8 * khalf, j = q * 16 + c;
      float hv = h0 ? (float)h0[(size_t)(b0 + m) * H_ + j] : 0.0f;
      hreg[q][v] = hv;
      shw[m * H_ + j] = (_Float16)hv;
    }

  // Software pipeline for the input fragments: the global load for step t+1 is
  // issued while step t computes, keeping HBM/L2 latency off the recurrence.
  const _Float16* xbase = in + (size_t)(b0 + c) * in_bstride;
  v16h ax0 = loadA(xbase, 0, khalf);   // t = 0
  v16h ax1 = loadA(xbase, 1, khalf);

  for (int t = 0; t < T_; ++t) {
    // Prefetch next timestep's x fragment (uniform clamped index, no branch).
    const int tn = (t + 1 < T_) ? (t + 1) : t;
    const _Float16* xrow_n = xbase + (size_t)tn * in_tstride;
    v16h ax0n = loadA(xrow_n, 0, khalf);
    v16h ax1n = loadA(xrow_n, 1, khalf);

    // Recurrent A fragments from wave-private LDS (true serial dependence).
    const _Float16* hrow = shw + c * H_;
    v16h ah0 = loadA(hrow, 0, khalf);
    v16h ah1 = loadA(hrow, 1, khalf);

    // r (tiles 0-3) and z (tiles 4-7): x-proj and h-proj share one accumulator.
    v8f accrz[8];
#pragma unroll
    for (int nt = 0; nt < 8; ++nt) {
      const _Float16* wi = sWih + (nt * 16 + c) * H_;
      const _Float16* wh = sWhh + (nt * 16 + c) * H_;
      v8f a = {0, 0, 0, 0, 0, 0, 0, 0};
      a = wmma(ax0, loadB(wi, 0, khalf), a);
      a = wmma(ax1, loadB(wi, 1, khalf), a);
      a = wmma(ah0, loadB(wh, 0, khalf), a);
      a = wmma(ah1, loadB(wh, 1, khalf), a);
      accrz[nt] = a;
    }
    // n (tiles 8-11): keep x-proj and h-proj separate (r gates only the h part).
    v8f accnx[4], accnh[4];
#pragma unroll
    for (int q = 0; q < 4; ++q) {
      const int nt = 8 + q;
      const _Float16* wi = sWih + (nt * 16 + c) * H_;
      const _Float16* wh = sWhh + (nt * 16 + c) * H_;
      v8f ax = {0, 0, 0, 0, 0, 0, 0, 0};
      ax = wmma(ax0, loadB(wi, 0, khalf), ax);
      ax = wmma(ax1, loadB(wi, 1, khalf), ax);
      v8f ah = {0, 0, 0, 0, 0, 0, 0, 0};
      ah = wmma(ah0, loadB(wh, 0, khalf), ah);
      ah = wmma(ah1, loadB(wh, 1, khalf), ah);
      accnx[q] = ax;
      accnh[q] = ah;
    }

    // Element-wise GRU cell, h update, LDS mirror + sequence output (f16).
#pragma unroll
    for (int q = 0; q < 4; ++q) {
#pragma unroll
      for (int v = 0; v < 8; ++v) {
        float r  = sigm(accrz[q][v] + brz_r[q]);
        float z  = sigm(accrz[4 + q][v] + brz_z[q]);
        float n  = tanh_fast(accnx[q][v] + b_in[q] + r * (accnh[q][v] + b_hn[q]));
        float hn = (1.0f - z) * n + z * hreg[q][v];
        hreg[q][v] = hn;
        const int m = v + 8 * khalf, j = q * 16 + c;
        _Float16 hh = (_Float16)hn;
        shw[m * H_ + j] = hh;                                    // feeds next step's A_h
        yseq[((size_t)(b0 + m) * T_ + t) * H_ + j] = hh;         // layer output seq
      }
    }

    // Rotate the pipelined x fragments.
    ax0 = ax0n;
    ax1 = ax1n;
    // wave-private LDS region: DScnt ordering within the wave is sufficient.
  }

#pragma unroll
  for (int q = 0; q < 4; ++q)
#pragma unroll
    for (int v = 0; v < 8; ++v) {
      const int m = v + 8 * khalf, j = q * 16 + c;
      hT[(size_t)(b0 + m) * H_ + j] = (_Float16)hreg[q][v];
    }
}

// FC: out[BT,64] = seq[BT,64] @ W^T + b, WMMA over M-tiles of 16 rows.
__global__ __launch_bounds__(128)
void fc_kernel(const _Float16* __restrict__ seq, const float* __restrict__ W,
               const float* __restrict__ bias, float* __restrict__ out)
{
  __shared__ _Float16 sW[F_ * H_];
  __shared__ float    sb[F_];
  const int tid = threadIdx.x;
  for (int i = tid; i < F_ * H_; i += 128) sW[i] = (_Float16)W[i];
  if (tid < F_) sb[tid] = bias[tid];
  __syncthreads();

  const int lane = tid & 31, wave = tid >> 5;
  const int c = lane & 15, khalf = lane >> 4;
  const size_t m0 = ((size_t)blockIdx.x * 4 + wave) * 16;

  const _Float16* arow = seq + (m0 + c) * H_;
  v16h a0 = loadA(arow, 0, khalf);
  v16h a1 = loadA(arow, 1, khalf);

#pragma unroll
  for (int nt = 0; nt < 4; ++nt) {
    const _Float16* wr = sW + (nt * 16 + c) * H_;
    v8f acc = {0, 0, 0, 0, 0, 0, 0, 0};
    acc = wmma(a0, loadB(wr, 0, khalf), acc);
    acc = wmma(a1, loadB(wr, 1, khalf), acc);
    const float bb = sb[nt * 16 + c];
#pragma unroll
    for (int v = 0; v < 8; ++v) {
      const int m = v + 8 * khalf;
      out[(m0 + m) * F_ + nt * 16 + c] = acc[v] + bb;
    }
  }
}

__global__ void f32_to_f16_kernel(const float* __restrict__ src, _Float16* __restrict__ dst, size_t n) {
  size_t i = (size_t)blockIdx.x * blockDim.x + threadIdx.x;
  const size_t stride = (size_t)gridDim.x * blockDim.x;
  for (; i < n; i += stride) dst[i] = (_Float16)src[i];
}

extern "C" void kernel_launch(void* const* d_in, const int* in_sizes, int n_in,
                              void* d_out, int out_size, void* d_ws, size_t ws_size,
                              hipStream_t stream)
{
  (void)in_sizes; (void)n_in; (void)out_size; (void)ws_size;
  const float* x     = (const float*)d_in[0];
  const float* eWih0 = (const float*)d_in[1];
  const float* eWhh0 = (const float*)d_in[2];
  const float* ebih0 = (const float*)d_in[3];
  const float* ebhh0 = (const float*)d_in[4];
  const float* eWih1 = (const float*)d_in[5];
  const float* eWhh1 = (const float*)d_in[6];
  const float* ebih1 = (const float*)d_in[7];
  const float* ebhh1 = (const float*)d_in[8];
  const float* dWih0 = (const float*)d_in[9];
  const float* dWhh0 = (const float*)d_in[10];
  const float* dbih0 = (const float*)d_in[11];
  const float* dbhh0 = (const float*)d_in[12];
  const float* dWih1 = (const float*)d_in[13];
  const float* dWhh1 = (const float*)d_in[14];
  const float* dbih1 = (const float*)d_in[15];
  const float* dbhh1 = (const float*)d_in[16];
  const float* fcW   = (const float*)d_in[17];
  const float* fcb   = (const float*)d_in[18];

  const size_t SEQ = (size_t)B_ * T_ * H_;          // 52,428,800 elems
  char* ws = (char*)d_ws;
  _Float16* bufA = (_Float16*)ws;                   // f16 seq ping
  _Float16* bufB = (_Float16*)(ws + SEQ * 2);       // f16 seq pong
  _Float16* hts  = (_Float16*)(ws + 2 * SEQ * 2);   // 4 x [B,H] f16
  _Float16* h_e0 = hts + 0 * (size_t)B_ * H_;
  _Float16* h_e1 = hts + 1 * (size_t)B_ * H_;
  _Float16* h_d0 = hts + 2 * (size_t)B_ * H_;
  _Float16* h_d1 = hts + 3 * (size_t)B_ * H_;

  // x (f32) -> f16
  f32_to_f16_kernel<<<2048, 256, 0, stream>>>(x, bufA, SEQ);

  const dim3 gb(B_ / (GRU_WAVES * 16));  // 256 blocks, 512 waves total
  const dim3 tb(GRU_WAVES * 32);
  const long long SB = (long long)T_ * H_, ST = (long long)H_;

  // Encoder layer 0: x -> bufB, hT -> h_e0
  gru_layer_kernel<<<gb, tb, 0, stream>>>(bufA, SB, ST, eWih0, eWhh0, ebih0, ebhh0,
                                          nullptr, bufB, h_e0);
  // Encoder layer 1: bufB -> bufA (seq unused downstream), hT -> h_e1 (= context)
  gru_layer_kernel<<<gb, tb, 0, stream>>>(bufB, SB, ST, eWih1, eWhh1, ebih1, ebhh1,
                                          nullptr, bufA, h_e1);
  // Decoder layer 0: input = context broadcast over T (tstride=0), h0 = h_e0
  gru_layer_kernel<<<gb, tb, 0, stream>>>(h_e1, (long long)H_, 0, dWih0, dWhh0, dbih0, dbhh0,
                                          h_e0, bufB, h_d0);
  // Decoder layer 1: bufB -> bufA, h0 = h_e1
  gru_layer_kernel<<<gb, tb, 0, stream>>>(bufB, SB, ST, dWih1, dWhh1, dbih1, dbhh1,
                                          h_e1, bufA, h_d1);
  // FC: bufA [BT,64] f16 -> d_out [B,T,F] f32
  fc_kernel<<<(B_ * T_) / 64, 128, 0, stream>>>(bufA, fcW, fcb, (float*)d_out);
}
